// RMB_22342419874203
// MI455X (gfx1250) — matrix operand order repaired and checked by reading, not compiled, not run on hardware
//
#include <hip/hip_runtime.h>
#include <math.h>
#include <stdint.h>

// ---------------------------------------------------------------------------
// MI455X / gfx1250 implementation of the SASSM + MoE block (compile-only).
// wave32, WMMA bf16 GEMMs (16x64 per wave), async-to-LDS staging (ASYNCcnt),
// chunked-parallel selective scan, in-LDS FFT.
// ---------------------------------------------------------------------------

#define LL      4096      // H*W
#define HH      64
#define WW2     64
#define DM      128       // d_model
#define DI      256       // d_inner
#define NS      16        // d_state
#define KK      4         // scan directions
#define NCHUNK  64
#define CHLEN   64        // NCHUNK*CHLEN == LL

typedef __attribute__((ext_vector_type(16))) __bf16 v16bf;
typedef __attribute__((ext_vector_type(8)))  float  v8f;

// --------------------- async global->LDS copy helpers ----------------------
// Issues one b128 async DMA (tracked by ASYNCcnt). lds_off: wave-relative LDS
// byte offset (low 32 bits of the flat shared address); gaddr: global address.
__device__ __forceinline__ void async_b128(uint32_t lds_off, uint64_t gaddr) {
  asm volatile("global_load_async_to_lds_b128 %0, %1, off"
               :: "v"(lds_off), "v"(gaddr) : "memory");
}
__device__ __forceinline__ void wait_async0() {
  asm volatile("s_wait_asynccnt 0x0" ::: "memory");
}

// ------------------------------- LayerNorm ---------------------------------
// one wave per row of 128; 4 rows per 128-thread block
__global__ void k_ln128(const float* __restrict__ x, const float* __restrict__ w,
                        const float* __restrict__ b, float* outf, __bf16* outb, int rows) {
  int wave = threadIdx.x >> 5, lane = threadIdx.x & 31;
  int row = blockIdx.x * 4 + wave;
  if (row >= rows) return;
  const float* xr = x + (size_t)row * DM;
  float4 v = ((const float4*)xr)[lane];
  float s = v.x + v.y + v.z + v.w;
  for (int o = 16; o > 0; o >>= 1) s += __shfl_xor(s, o, 32);
  float mu = s * (1.0f / DM);
  float d0 = v.x - mu, d1 = v.y - mu, d2 = v.z - mu, d3 = v.w - mu;
  float q = d0*d0 + d1*d1 + d2*d2 + d3*d3;
  for (int o = 16; o > 0; o >>= 1) q += __shfl_xor(q, o, 32);
  float rstd = rsqrtf(q * (1.0f / DM) + 1e-5f);
  float e[4] = {d0, d1, d2, d3};
  for (int i = 0; i < 4; i++) {
    int c = lane * 4 + i;
    float o = e[i] * rstd * w[c] + b[c];
    if (outf) outf[(size_t)row * DM + c] = o;
    outb[(size_t)row * DM + c] = (__bf16)o;
  }
}

// out_norm(256) fused with * silu(z); z = xz[l][256+c]
__global__ void k_ln_gate(const float* __restrict__ ysum, const float* __restrict__ xz,
                          const float* __restrict__ w, const float* __restrict__ b,
                          __bf16* __restrict__ gated) {
  int wave = threadIdx.x >> 5, lane = threadIdx.x & 31;
  int row = blockIdx.x * 4 + wave;
  if (row >= LL) return;
  const float* xr = ysum + (size_t)row * DI;
  float4 v0 = ((const float4*)xr)[lane * 2];
  float4 v1 = ((const float4*)xr)[lane * 2 + 1];
  float s = v0.x+v0.y+v0.z+v0.w + v1.x+v1.y+v1.z+v1.w;
  for (int o = 16; o > 0; o >>= 1) s += __shfl_xor(s, o, 32);
  float mu = s * (1.0f / DI);
  float e[8] = {v0.x-mu, v0.y-mu, v0.z-mu, v0.w-mu, v1.x-mu, v1.y-mu, v1.z-mu, v1.w-mu};
  float q = 0.f;
  for (int i = 0; i < 8; i++) q += e[i]*e[i];
  for (int o = 16; o > 0; o >>= 1) q += __shfl_xor(q, o, 32);
  float rstd = rsqrtf(q * (1.0f / DI) + 1e-5f);
  const float* zr = xz + (size_t)row * (2*DI) + DI;
  for (int i = 0; i < 8; i++) {
    int c = lane * 8 + i;
    float o = e[i] * rstd * w[c] + b[c];
    float z = zr[c];
    o *= z / (1.f + expf(-z));              // silu(z)
    gated[(size_t)row * DI + c] = (__bf16)o;
  }
}

// --------------------------- f32 -> bf16 converts --------------------------
__global__ void k_cvt_bf16(const float* __restrict__ src, __bf16* __restrict__ dst, int n) {
  int i = blockIdx.x * blockDim.x + threadIdx.x;
  if (i < n) dst[i] = (__bf16)src[i];
}
// x_proj_w (4,40,256) -> padded (4,48,256) bf16, rows 40..47 zero
__global__ void k_pad_xproj(const float* __restrict__ src, __bf16* __restrict__ dst) {
  int i = blockIdx.x * blockDim.x + threadIdx.x;        // 4*48*256
  if (i >= KK*48*DI) return;
  int k = i / (48*DI), r = i % (48*DI), c = r / DI, d = r % DI;
  dst[i] = (c < 40) ? (__bf16)src[(k*40 + c)*DI + d] : (__bf16)0.f;
}

// ------------------------------ WMMA GEMM ----------------------------------
// Cout[M,N] = A[M,K] * Bw[N,K]^T (+bias[N]) (+resid[M,N]).
// One wave per 16x(16*NT) strip: A fragment reused NT times, NT independent
// WMMAs per K-step keep the XDL pipe fed while fragments stream from L2.
template<int NT>
__global__ void k_gemm_bf16(const __bf16* __restrict__ A, const __bf16* __restrict__ Bw,
                            const float* __restrict__ bias, const float* __restrict__ resid,
                            float* __restrict__ Cout, int M, int N, int Kd) {
  int wave = threadIdx.x >> 5, lane = threadIdx.x & 31;
  int tilesNg = N / (16 * NT);
  int tile = blockIdx.x * (blockDim.x >> 5) + wave;
  if (tile >= (M >> 4) * tilesNg) return;               // wave-uniform, EXEC stays all-1
  int mB = (tile / tilesNg) << 4, nB = (tile % tilesNg) * 16 * NT;
  int m = lane & 15, hi = lane >> 4;
  v8f acc[NT];
  for (int t = 0; t < NT; t++)
    for (int i = 0; i < 8; i++) acc[t][i] = 0.f;
  const __bf16* Arow = A + (size_t)(mB + m) * Kd;
  const __bf16* Brow[NT];
#pragma unroll
  for (int t = 0; t < NT; t++) Brow[t] = Bw + (size_t)(nB + t*16 + m) * Kd;
  for (int k0 = 0; k0 < Kd; k0 += 32) {
    union { uint4 u[2]; v16bf v; } af;
    af.u[0] = *(const uint4*)(Arow + k0 + hi*8);         // A: K = k0 + hi*8 + j
    af.u[1] = *(const uint4*)(Arow + k0 + 16 + hi*8);    //    K = k0+16+hi*8 + j
    if (k0 + 32 < Kd)                                    // gfx1250 global_prefetch_b8
      __builtin_prefetch((const void*)(Arow + k0 + 32), 0, 3);
#pragma unroll
    for (int t = 0; t < NT; t++) {
      union { uint4 u[2]; v16bf v; } bf_;
      bf_.u[0] = *(const uint4*)(Brow[t] + k0 + hi*8);   // B: K = k0 + hi*8 + j
      bf_.u[1] = *(const uint4*)(Brow[t] + k0 + 16 + hi*8);
      acc[t] = __builtin_amdgcn_wmma_f32_16x16x32_bf16(false, af.v, false, bf_.v,
                                                       (short)0, acc[t], false, false);
    }
  }
#pragma unroll
  for (int t = 0; t < NT; t++) {
    int col = nB + t*16 + m;
    float bval = bias ? bias[col] : 0.f;
    for (int v = 0; v < 8; v++) {                        // acc[v] -> row v+8*hi, col
      size_t idx = (size_t)(mB + v + 8*hi) * N + col;
      float o = acc[t][v] + bval;
      if (resid) o += resid[idx];
      Cout[idx] = o;
    }
  }
}

// -------------------- depthwise 3x3 + SiLU (groups=256) --------------------
__global__ void k_dwconv_silu(const float* __restrict__ xz, const float* __restrict__ wgt,
                              const float* __restrict__ bias, float* __restrict__ xfeat,
                              float* __restrict__ xfeat_t) {
  int idx = blockIdx.x * blockDim.x + threadIdx.x;       // LL*256
  int l = idx >> 8, c = idx & 255;
  int h = l >> 6, w = l & 63;
  float s = bias[c];
  for (int dy = -1; dy <= 1; dy++)
    for (int dx = -1; dx <= 1; dx++) {
      int hh = h + dy, ww = w + dx;
      if (hh >= 0 && hh < HH && ww >= 0 && ww < WW2)
        s += xz[(size_t)(hh*WW2 + ww) * (2*DI) + c] * wgt[c*9 + (dy+1)*3 + (dx+1)];
    }
  float f = s / (1.f + expf(-s));
  xfeat[(size_t)c * LL + l] = f;
  xfeat_t[(size_t)l * DI + c] = f;
}

// --------------------- phase-only FFT reconstruction -----------------------
__device__ __forceinline__ void fft_stages(float* sre, float* sim, int t, int ax) {
  for (int s = 0; s < 6; s++) {
    int half = 1 << s;
    for (int e = 0; e < 8; e++) {
      int p = t * 8 + e;                                 // 2048 butterflies / stage
      int tr = ax ? (p & 63) : (p >> 5);
      int j  = ax ? (p >> 6) : (p & 31);
      int j2 = j & (half - 1);
      int pos = ((j >> s) << (s + 1)) + j2;
      int i1 = ax ? (pos * 64 + tr)        : (tr * 64 + pos);
      int i2 = ax ? ((pos + half)*64 + tr) : (tr * 64 + pos + half);
      float ang = -6.283185307179586f * (float)j2 / (float)(2 * half);
      float sn, cs; __sincosf(ang, &sn, &cs);
      float xr = sre[i2], xi = sim[i2];
      float tre = xr*cs - xi*sn, tim = xr*sn + xi*cs;
      float ar = sre[i1], ai = sim[i1];
      sre[i1] = ar + tre; sim[i1] = ai + tim;
      sre[i2] = ar - tre; sim[i2] = ai - tim;
    }
    __syncthreads();
  }
}
__device__ __forceinline__ void bitrev_axis(float* sre, float* sim, int t, int ax) {
  for (int e = 0; e < 16; e++) {
    int i = t * 16 + e;
    int r = i >> 6, cc = i & 63;
    int a = ax ? r : cc;
    int b = __brev((unsigned)a) >> 26;
    if (a < b) {
      int i2 = ax ? (b * 64 + cc) : (r * 64 + b);
      float tr = sre[i]; sre[i] = sre[i2]; sre[i2] = tr;
      float ti = sim[i]; sim[i] = sim[i2]; sim[i2] = ti;
    }
  }
  __syncthreads();
}
__global__ void k_fft_phase(const float* __restrict__ xfeat, float* __restrict__ recon_t) {
  __shared__ float sre[4096];
  __shared__ float sim[4096];
  int c = blockIdx.x, t = threadIdx.x;
  const float* src = xfeat + (size_t)c * LL;
  for (int e = 0; e < 16; e++) {                         // load, col-bitrev read
    int i = t * 16 + e;
    sre[i] = src[(i & ~63) | (__brev((unsigned)(i & 63)) >> 26)];
    sim[i] = 0.f;
  }
  __syncthreads();
  fft_stages(sre, sim, t, 0);                            // FFT rows
  bitrev_axis(sre, sim, t, 1);
  fft_stages(sre, sim, t, 1);                            // FFT cols: full forward FFT
  for (int e = 0; e < 16; e++) {                         // phase-only + conj (for inverse)
    int i = t * 16 + e;
    float re = sre[i], im = sim[i];
    float mag = sqrtf(re*re + im*im);
    if (mag > 1e-20f) { re /= mag; im /= mag; } else { re = 1.f; im = 0.f; }
    sre[i] = re; sim[i] = -im;
  }
  __syncthreads();
  bitrev_axis(sre, sim, t, 0);
  fft_stages(sre, sim, t, 0);
  bitrev_axis(sre, sim, t, 1);
  fft_stages(sre, sim, t, 1);
  for (int e = 0; e < 16; e++) {                         // recon = Re * 1/64 (ortho pair)
    int i = t * 16 + e;
    recon_t[(size_t)i * DI + c] = sre[i] * (1.0f / 64.0f);
  }
}

// ----------------------- pe1/pe2 -> struct (B,4,H,W) -----------------------
__global__ void k_pe(const float* __restrict__ recon_t, const float* __restrict__ w1,
                     const float* __restrict__ b1, const float* __restrict__ w2,
                     const float* __restrict__ b2, float* __restrict__ struct4) {
  __shared__ float rrow[DI];
  __shared__ float tbuf[64];
  int l = blockIdx.x, o = threadIdx.x;
  // async DMA the 1KB recon row into LDS (64 lanes x b128)
  async_b128((uint32_t)(uintptr_t)(&rrow[0]) + (uint32_t)o * 16u,
             (uint64_t)(uintptr_t)(recon_t + (size_t)l * DI) + (uint64_t)o * 16u);
  float s = b1[o];
  wait_async0();
  __syncthreads();
  for (int c2 = 0; c2 < DI; c2++) s += rrow[c2] * w1[o*DI + c2];
  tbuf[o] = s > 0.f ? s : 0.2f * s;                      // leaky 0.2
  __syncthreads();
  if (o < 4) {
    float a = b2[o];
    for (int i = 0; i < 64; i++) a += tbuf[i] * w2[o*64 + i];
    struct4[o*LL + l] = 1.f / (1.f + expf(-a));          // sigmoid
  }
}

// ---------------------- illumination gate -> ssa ---------------------------
__global__ void k_illum(const float* __restrict__ xfeat_t, float* __restrict__ illum) {
  int l = blockIdx.x * blockDim.x + threadIdx.x;
  if (l >= LL) return;
  float s = 0.f;
  for (int c = 0; c < DI; c++) s += xfeat_t[(size_t)l * DI + c];
  illum[l] = s * (1.0f / DI);
}
__global__ void k_ig(const float* __restrict__ illum, const float* __restrict__ w1,
                     const float* __restrict__ b1, const float* __restrict__ w2,
                     const float* __restrict__ b2, float* __restrict__ ssa) {
  __shared__ float gbuf[64];
  __shared__ float e4[4];
  int l = blockIdx.x, o = threadIdx.x;
  int h = l >> 6, w = l & 63;
  float s = b1[o];
  for (int dy = -1; dy <= 1; dy++)
    for (int dx = -1; dx <= 1; dx++) {
      int hh = h + dy, ww = w + dx;
      if (hh >= 0 && hh < HH && ww >= 0 && ww < WW2)
        s += illum[hh*WW2 + ww] * w1[o*9 + (dy+1)*3 + (dx+1)];
    }
  gbuf[o] = s > 0.f ? s : 0.2f * s;
  __syncthreads();
  if (o < 4) {
    float a = b2[o];
    for (int i = 0; i < 64; i++) a += gbuf[i] * w2[o*64 + i];
    e4[o] = a;
  }
  __syncthreads();
  if (o < 4) {
    float m = fmaxf(fmaxf(e4[0], e4[1]), fmaxf(e4[2], e4[3]));
    float z = expf(e4[0]-m) + expf(e4[1]-m) + expf(e4[2]-m) + expf(e4[3]-m);
    ssa[o*LL + l] = expf(e4[o]-m) / z;
  }
}

// ------------------- gather xs (scan-ordered, bf16, [k][l][d]) -------------
__global__ void k_gather_xs(const float* __restrict__ xfeat_t, const int* __restrict__ sid,
                            __bf16* __restrict__ xs_a) {
  int idx = blockIdx.x * blockDim.x + threadIdx.x;       // K*LL*DI
  int k = idx >> 20, rem = idx & ((1 << 20) - 1);
  int l = rem >> 8, d = rem & 255;
  xs_a[idx] = (__bf16)xfeat_t[(size_t)sid[k*LL + l] * DI + d];
}

// -------------------- chunked-parallel selective scan ----------------------
// pass A (FINAL=false): per-chunk (prod dA, local end state) -> chunkstate
// pass C (FINAL=true):  replay with carry-in, emit y = sum_n h*C + D*u
// x_dbl chunk (12KB contiguous) is DMA'd to LDS asynchronously; the struct
// gate g is folded in at consumption: dt=softplus(b+g<w,r>), B*=g, y*=g.
template<bool FINAL>
__global__ void k_scan(const float* __restrict__ xdbl, const float* __restrict__ struct4,
                       const int* __restrict__ scan_ids, const float* __restrict__ xfeat_t,
                       const float* __restrict__ dt_w, const float* __restrict__ dt_b,
                       const float* __restrict__ A_logs, const float* __restrict__ Ds,
                       float* __restrict__ chunkstate, float* __restrict__ out_y) {
  __shared__ float xb[CHLEN * 48];
  __shared__ float sgate[CHLEN];
  __shared__ int sids[CHLEN];
  int chunk = blockIdx.x, k = blockIdx.y, d = threadIdx.x;
  int l0 = chunk * CHLEN;
  // kick off async DMA of the x_dbl chunk (3072 floats contiguous, 3 b128/lane)
  {
    uint32_t lbase = (uint32_t)(uintptr_t)(&xb[0]);
    uint64_t gbase = (uint64_t)(uintptr_t)(xdbl + ((size_t)k*LL + l0) * 48);
#pragma unroll
    for (int i = 0; i < 3; i++)
      async_b128(lbase + (uint32_t)(threadIdx.x + i*256) * 16u,
                 gbase + (uint64_t)(threadIdx.x + i*256) * 16u);
  }
  if (threadIdx.x < CHLEN) {
    sids[threadIdx.x] = scan_ids[k*LL + l0 + threadIdx.x];
    sgate[threadIdx.x] = struct4[k*LL + l0 + threadIdx.x];
  }
  // per-lane constants overlap with the in-flight DMA
  int kd = k * DI + d;
  float dtw[8];
  for (int r = 0; r < 8; r++) dtw[r] = dt_w[kd*8 + r];
  float dtb = dt_b[kd];
  float Av[NS], hst[NS], P[NS];
  float* cs = chunkstate + ((size_t)(k*NCHUNK + chunk) * DI + d) * 32;
  for (int n = 0; n < NS; n++) {
    Av[n] = -__expf(A_logs[kd*NS + n]);
    hst[n] = FINAL ? cs[n] : 0.f;                        // pass B wrote carry-in at [0..16)
    P[n] = 1.f;
  }
  float Dv = Ds[kd];
  wait_async0();
  __syncthreads();
  for (int i = 0; i < CHLEN; i++) {
    const float* row = xb + i * 48;
    float g = sgate[i];
    float dot = 0.f;
    for (int r = 0; r < 8; r++) dot += dtw[r] * row[r];  // fused dt_proj
    float dtr = dtb + g * dot;
    float dt = dtr > 20.f ? dtr : log1pf(__expf(dtr));   // softplus
    float u = xfeat_t[(size_t)sids[i] * DI + d];
    float dug = dt * u * g;
    float y = 0.f;
    for (int n = 0; n < NS; n++) {
      float dA = __expf(dt * Av[n]);
      hst[n] = hst[n] * dA + dug * row[8 + n];
      if (!FINAL) P[n] *= dA;
      else        y += hst[n] * row[24 + n];
    }
    if (FINAL) out_y[((size_t)k*LL + l0 + i) * DI + d] = g * y + Dv * u;
  }
  if (!FINAL)
    for (int n = 0; n < NS; n++) { cs[n] = P[n]; cs[16 + n] = hst[n]; }
}
// pass B: prefix over chunks, one lane per (k,d,n)
__global__ void k_scan_prefix(float* __restrict__ chunkstate) {
  int idx = blockIdx.x * blockDim.x + threadIdx.x;       // K*DI*NS = 16384
  int k = idx / (DI*NS), rem = idx % (DI*NS), d = rem >> 4, n = rem & 15;
  float carry = 0.f;
  for (int chunk = 0; chunk < NCHUNK; chunk++) {
    float* cs = chunkstate + ((size_t)(k*NCHUNK + chunk) * DI + d) * 32;
    float Pv = cs[n], hl = cs[16 + n];
    cs[n] = carry;                                       // carry-in for this chunk
    carry = Pv * carry + hl;
  }
}

// -------------- inverse-scan gather + ssa weight + sum over k --------------
__global__ void k_inv_combine(const float* __restrict__ out_y, const int* __restrict__ inv_ids,
                              const float* __restrict__ ssa, float* __restrict__ ysum) {
  int idx = blockIdx.x * blockDim.x + threadIdx.x;       // LL*DI
  int i = idx >> 8, d = idx & 255;
  float acc = 0.f;
  for (int k = 0; k < KK; k++) {
    int j = inv_ids[k*LL + i];
    acc += out_y[((size_t)k*LL + j) * DI + d] * ssa[k*LL + i];
  }
  ysum[(size_t)i * DI + d] = acc;
}

// ------------------------------- MoE pieces --------------------------------
__global__ void k_It(const float* __restrict__ xn2, float* __restrict__ It) {
  int l = blockIdx.x * blockDim.x + threadIdx.x;
  if (l >= LL) return;
  float s = 0.f;
  for (int c = 0; c < DM; c++) s += xn2[(size_t)l * DM + c];
  It[l] = s * (1.0f / DM);
}
__device__ __forceinline__ float gelu_exact(float x) {
  return 0.5f * x * (1.f + erff(x * 0.70710678118654752f));
}
__global__ void k_route(const float* __restrict__ It, const float* __restrict__ w1,
                        const float* __restrict__ b1, const float* __restrict__ w2,
                        const float* __restrict__ b2, float* __restrict__ alpha) {
  int l = blockIdx.x * blockDim.x + threadIdx.x;
  if (l >= LL) return;
  int h = l >> 6, w = l & 63;
  float g[8];
  for (int o = 0; o < 8; o++) {
    float s = b1[o];
    for (int dy = -1; dy <= 1; dy++)
      for (int dx = -1; dx <= 1; dx++) {
        int hh = h + dy, ww = w + dx;
        if (hh >= 0 && hh < HH && ww >= 0 && ww < WW2)
          s += It[hh*WW2 + ww] * w1[o*9 + (dy+1)*3 + (dx+1)];
      }
    g[o] = gelu_exact(s);
  }
  float a[3];
  for (int e = 0; e < 3; e++) {
    float s = b2[e];
    for (int o = 0; o < 8; o++) s += g[o] * w2[e*8 + o];
    a[e] = s;
  }
  float m = fmaxf(a[0], fmaxf(a[1], a[2]));
  float z = expf(a[0]-m) + expf(a[1]-m) + expf(a[2]-m);
  for (int e = 0; e < 3; e++) alpha[e*LL + l] = expf(a[e]-m) / z;
}
// Fout = F + (sum_e alpha_e * dwconv_e(F)) * F   (3 experts fused, no Fs buffer)
__global__ void k_moe_combine(const float* __restrict__ F, const float* __restrict__ alpha,
                              const float* __restrict__ e0w, const float* __restrict__ e0b,
                              const float* __restrict__ e1w, const float* __restrict__ e1b,
                              const float* __restrict__ e2w, const float* __restrict__ e2b,
                              float* __restrict__ Fout) {
  int idx = blockIdx.x * blockDim.x + threadIdx.x;       // LL*256
  int l = idx >> 8, c = idx & 255;
  int h = l >> 6, w = l & 63;
  float s0 = e0b[c], s1 = e1b[c], s2 = e2b[c];
  for (int dy = -1; dy <= 1; dy++)
    for (int dx = -1; dx <= 1; dx++) {
      int hh = h + dy, ww = w + dx;
      if (hh >= 0 && hh < HH && ww >= 0 && ww < WW2) {
        float f = F[(size_t)(hh*WW2 + ww) * DI + c];
        int tap = c*9 + (dy+1)*3 + (dx+1);
        s0 += f * e0w[tap]; s1 += f * e1w[tap]; s2 += f * e2w[tap];
      }
    }
  float Fv = F[(size_t)l * DI + c];
  float mix = alpha[0*LL + l]*s0 + alpha[1*LL + l]*s1 + alpha[2*LL + l]*s2;
  Fout[(size_t)l * DI + c] = Fv + mix * Fv;
}
__global__ void k_dcg(const float* __restrict__ Fout, const float* __restrict__ dcw,
                      const float* __restrict__ dcb, __bf16* __restrict__ dcg) {
  int idx = blockIdx.x * blockDim.x + threadIdx.x;       // LL*256
  int l = idx >> 8, c = idx & 255;
  int h = l >> 6, w = l & 63;
  float s = dcb[c];
  for (int dy = -1; dy <= 1; dy++)
    for (int dx = -1; dx <= 1; dx++) {
      int hh = h + dy, ww = w + dx;
      if (hh >= 0 && hh < HH && ww >= 0 && ww < WW2)
        s += Fout[(size_t)(hh*WW2 + ww) * DI + c] * dcw[c*9 + (dy+1)*3 + (dx+1)];
    }
  dcg[(size_t)l * DI + c] = (__bf16)gelu_exact(s);
}

// ------------------------------- launcher ----------------------------------
extern "C" void kernel_launch(void* const* d_in, const int* in_sizes, int n_in,
                              void* d_out, int out_size, void* d_ws, size_t ws_size,
                              hipStream_t stream) {
  (void)in_sizes; (void)n_in; (void)out_size; (void)ws_size;
  // input order: x, params (dict insertion order), scan_ids, inv_ids
  const float* x        = (const float*)d_in[0];
  const float* ln1_w    = (const float*)d_in[1];
  const float* ln1_b    = (const float*)d_in[2];
  const float* ln2_w    = (const float*)d_in[3];
  const float* ln2_b    = (const float*)d_in[4];
  const float* in_proj_w= (const float*)d_in[5];
  const float* conv2d_w = (const float*)d_in[6];
  const float* conv2d_b = (const float*)d_in[7];
  const float* pe1_w    = (const float*)d_in[8];
  const float* pe1_b    = (const float*)d_in[9];
  const float* pe2_w    = (const float*)d_in[10];
  const float* pe2_b    = (const float*)d_in[11];
  const float* ig1_w    = (const float*)d_in[12];
  const float* ig1_b    = (const float*)d_in[13];
  const float* ig2_w    = (const float*)d_in[14];
  const float* ig2_b    = (const float*)d_in[15];
  const float* x_proj_w = (const float*)d_in[16];
  const float* dt_proj_w= (const float*)d_in[17];
  const float* dt_proj_b= (const float*)d_in[18];
  const float* A_logs   = (const float*)d_in[19];
  const float* Ds       = (const float*)d_in[20];
  const float* onorm_w  = (const float*)d_in[21];
  const float* onorm_b  = (const float*)d_in[22];
  const float* out_proj_w=(const float*)d_in[23];
  const float* phi_in_w = (const float*)d_in[24];
  const float* phi_in_b = (const float*)d_in[25];
  const float* exp0_w   = (const float*)d_in[26];
  const float* exp0_b   = (const float*)d_in[27];
  const float* exp1_w   = (const float*)d_in[28];
  const float* exp1_b   = (const float*)d_in[29];
  const float* exp2_w   = (const float*)d_in[30];
  const float* exp2_b   = (const float*)d_in[31];
  const float* rt1_w    = (const float*)d_in[32];
  const float* rt1_b    = (const float*)d_in[33];
  const float* rt2_w    = (const float*)d_in[34];
  const float* rt2_b    = (const float*)d_in[35];
  const float* dc_w     = (const float*)d_in[36];
  const float* dc_b     = (const float*)d_in[37];
  const float* phi_out_w= (const float*)d_in[38];
  const float* phi_out_b= (const float*)d_in[39];
  const int*   scan_ids = (const int*)d_in[40];
  const int*   inv_ids  = (const int*)d_in[41];

  // workspace carve (~78 MB, all L2-resident on MI455X's 192MB L2)
  char* base = (char*)d_ws;
  size_t off = 0;
  auto carve = [&](size_t bytes) -> void* {
    void* p = base + off;
    off += (bytes + 255) & ~(size_t)255;
    return p;
  };
  __bf16* xn1_bf    = (__bf16*)carve((size_t)LL*DM*2);
  __bf16* wb_inproj = (__bf16*)carve((size_t)2*DI*DM*2);
  __bf16* wb_xproj  = (__bf16*)carve((size_t)KK*48*DI*2);
  __bf16* wb_outproj= (__bf16*)carve((size_t)DM*DI*2);
  __bf16* wb_phiin  = (__bf16*)carve((size_t)DI*DM*2);
  __bf16* wb_phiout = (__bf16*)carve((size_t)DM*DI*2);
  float*  xz        = (float*)carve((size_t)LL*2*DI*4);
  float*  xfeat     = (float*)carve((size_t)DI*LL*4);
  float*  xfeat_t   = (float*)carve((size_t)LL*DI*4);
  float*  recon_t   = (float*)carve((size_t)LL*DI*4);
  float*  struct4   = (float*)carve((size_t)4*LL*4);
  float*  illum     = (float*)carve((size_t)LL*4);
  float*  ssa       = (float*)carve((size_t)4*LL*4);
  __bf16* xs_a      = (__bf16*)carve((size_t)KK*LL*DI*2);
  float*  xdbl      = (float*)carve((size_t)KK*LL*48*4);
  float*  chunkst   = (float*)carve((size_t)KK*NCHUNK*DI*32*4);
  float*  out_y     = (float*)carve((size_t)KK*LL*DI*4);
  float*  ysum      = (float*)carve((size_t)LL*DI*4);
  __bf16* gated_bf  = (__bf16*)carve((size_t)LL*DI*2);
  float*  hbuf      = (float*)carve((size_t)LL*DM*4);
  float*  xn2_f     = (float*)carve((size_t)LL*DM*4);
  __bf16* xn2_bf    = (__bf16*)carve((size_t)LL*DM*2);
  float*  Fbuf      = (float*)carve((size_t)LL*DI*4);
  float*  Itbuf     = (float*)carve((size_t)LL*4);
  float*  alpha     = (float*)carve((size_t)3*LL*4);
  float*  Foutb     = (float*)carve((size_t)LL*DI*4);
  __bf16* dcg_bf    = (__bf16*)carve((size_t)LL*DI*2);

  // weight converts (once per call; trivially cheap)
  k_cvt_bf16<<<(2*DI*DM+255)/256, 256, 0, stream>>>(in_proj_w, wb_inproj, 2*DI*DM);
  k_pad_xproj<<<(KK*48*DI+255)/256, 256, 0, stream>>>(x_proj_w, wb_xproj);
  k_cvt_bf16<<<(DM*DI+255)/256, 256, 0, stream>>>(out_proj_w, wb_outproj, DM*DI);
  k_cvt_bf16<<<(DI*DM+255)/256, 256, 0, stream>>>(phi_in_w, wb_phiin, DI*DM);
  k_cvt_bf16<<<(DM*DI+255)/256, 256, 0, stream>>>(phi_out_w, wb_phiout, DM*DI);

  // SASSM branch
  k_ln128<<<LL/4, 128, 0, stream>>>(x, ln1_w, ln1_b, nullptr, xn1_bf, LL);
  k_gemm_bf16<4><<<(LL/16)*(2*DI/64)/8, 256, 0, stream>>>(xn1_bf, wb_inproj, nullptr,
                                                          nullptr, xz, LL, 2*DI, DM);
  k_dwconv_silu<<<LL*DI/256, 256, 0, stream>>>(xz, conv2d_w, conv2d_b, xfeat, xfeat_t);
  k_fft_phase<<<DI, 256, 0, stream>>>(xfeat, recon_t);
  k_pe<<<LL, 64, 0, stream>>>(recon_t, pe1_w, pe1_b, pe2_w, pe2_b, struct4);
  k_illum<<<LL/256, 256, 0, stream>>>(xfeat_t, illum);
  k_ig<<<LL, 64, 0, stream>>>(illum, ig1_w, ig1_b, ig2_w, ig2_b, ssa);
  k_gather_xs<<<KK*LL*DI/256, 256, 0, stream>>>(xfeat_t, scan_ids, xs_a);
  for (int k = 0; k < KK; k++)
    k_gemm_bf16<3><<<(LL/16)*(48/48)/8, 256, 0, stream>>>(xs_a + (size_t)k*LL*DI,
                                                          wb_xproj + (size_t)k*48*DI,
                                                          nullptr, nullptr,
                                                          xdbl + (size_t)k*LL*48, LL, 48, DI);
  dim3 sg(NCHUNK, KK);
  k_scan<false><<<sg, 256, 0, stream>>>(xdbl, struct4, scan_ids, xfeat_t, dt_proj_w,
                                        dt_proj_b, A_logs, Ds, chunkst, out_y);
  k_scan_prefix<<<KK*DI*NS/256, 256, 0, stream>>>(chunkst);
  k_scan<true><<<sg, 256, 0, stream>>>(xdbl, struct4, scan_ids, xfeat_t, dt_proj_w,
                                       dt_proj_b, A_logs, Ds, chunkst, out_y);
  k_inv_combine<<<LL*DI/256, 256, 0, stream>>>(out_y, inv_ids, ssa, ysum);
  k_ln_gate<<<LL/4, 128, 0, stream>>>(ysum, xz, onorm_w, onorm_b, gated_bf);
  k_gemm_bf16<4><<<(LL/16)*(DM/64)/8, 256, 0, stream>>>(gated_bf, wb_outproj, nullptr, x,
                                                        hbuf, LL, DM, DI); // h = x + sassm

  // MoE branch
  k_ln128<<<LL/4, 128, 0, stream>>>(hbuf, ln2_w, ln2_b, xn2_f, xn2_bf, LL);
  k_gemm_bf16<4><<<(LL/16)*(DI/64)/8, 256, 0, stream>>>(xn2_bf, wb_phiin, phi_in_b,
                                                        nullptr, Fbuf, LL, DI, DM);
  k_It<<<LL/256, 256, 0, stream>>>(xn2_f, Itbuf);
  k_route<<<LL/256, 256, 0, stream>>>(Itbuf, rt1_w, rt1_b, rt2_w, rt2_b, alpha);
  k_moe_combine<<<LL*DI/256, 256, 0, stream>>>(Fbuf, alpha, exp0_w, exp0_b, exp1_w, exp1_b,
                                               exp2_w, exp2_b, Foutb);
  k_dcg<<<LL*DI/256, 256, 0, stream>>>(Foutb, dc_w, dc_b, dcg_bf);
  k_gemm_bf16<4><<<(LL/16)*(DM/64)/8, 256, 0, stream>>>(dcg_bf, wb_phiout, phi_out_b, hbuf,
                                                        (float*)d_out, LL, DM, DI);
}
